// FeedbackTransformer_29678224015623
// MI455X (gfx1250) — matrix operand order repaired and checked by reading, not compiled, not run on hardware
//
#include <hip/hip_runtime.h>
#include <hip/hip_bf16.h>

typedef __attribute__((ext_vector_type(16))) _Float16 v16h;
typedef __attribute__((ext_vector_type(8)))  _Float16 v8h;
typedef __attribute__((ext_vector_type(8)))  float    v8f;

namespace {
constexpr int Bb   = 8;
constexpr int Nn   = 4096;
constexpr int Ff   = 8192;   // fused candidate count N+M
constexpr int Cc   = 128;    // IN_CH
constexpr int Kn   = 16;     // KNN
constexpr int PH   = 64;     // POS_H
constexpr int AH   = 512;    // ATTN_H
constexpr int NPTS = Bb * Nn;          // 32768 query points
constexpr int PPB  = 16;               // points per block in WMMA kernels
constexpr float EPSf = 1e-5f;
}

#define DEVINL static __device__ __forceinline__

DEVINL v8f wmma_f16(v16h a, v16h b, v8f c) {
  // D = A(16x32 f16) * B(32x16 f16) + C(16x16 f32)
  return __builtin_amdgcn_wmma_f32_16x16x32_f16(
      /*neg_a=*/false, a, /*neg_b=*/false, b,
      /*c_mod=*/(short)0, c, /*reuse_a=*/false, /*reuse_b=*/false);
}

// A-fragment: row-major weight matrix; lane<16 holds K{kb..kb+15}
// (caller folds +16 into kb for lanes>=16).
DEVINL v16h load_a_frag(const _Float16* __restrict__ row, int kbase) {
  v8h lo = *(const v8h*)(row + kbase);
  v8h hi = *(const v8h*)(row + kbase + 8);
  return __builtin_shufflevector(lo, hi, 0,1,2,3,4,5,6,7,8,9,10,11,12,13,14,15);
}

// ---------------------------------------------------------------------------
// Kernel 0: weights fp32->fp16 into workspace; zero the stats buffers.
// ---------------------------------------------------------------------------
__global__ void k_prep(const float* __restrict__ pw2, const float* __restrict__ aw1,
                       const float* __restrict__ aw2,
                       _Float16* __restrict__ wpos2h, _Float16* __restrict__ wattn1h,
                       _Float16* __restrict__ wattn2h, float* __restrict__ stats) {
  int t = blockIdx.x * blockDim.x + threadIdx.x;
  if (t < Cc * PH) wpos2h[t]  = (_Float16)pw2[t];
  if (t < AH * Cc) wattn1h[t] = (_Float16)aw1[t];
  if (t < Cc * AH) wattn2h[t] = (_Float16)aw2[t];
  if (t < 2304)    stats[t]   = 0.f;
}

// ---------------------------------------------------------------------------
// Kernel 1: transpose fused feat (B,C,F)->(B,F,C) and gad (B,14,F)->(B,F,14)
// so neighbor gathers become contiguous 512B / 56B reads (L2-friendly).
// ---------------------------------------------------------------------------
__global__ void k_transpose(const float* __restrict__ feat, const float* __restrict__ featdb,
                            const float* __restrict__ gad,  const float* __restrict__ gaddb,
                            float* __restrict__ featT, float* __restrict__ gadT) {
  long e = (long)blockIdx.x * blockDim.x + threadIdx.x;
  const long totF = (long)Bb * Ff * Cc;
  if (e < totF) {
    int c = (int)(e % Cc); long r = e / Cc;
    int j = (int)(r % Ff); int b = (int)(r / Ff);
    featT[e] = (j < Nn) ? feat[((long)(b * Cc + c)) * Nn + j]
                        : featdb[((long)(b * Cc + c)) * Nn + (j - Nn)];
  }
  const long totG = (long)Bb * Ff * 14;
  if (e < totG) {
    int c = (int)(e % 14); long r = e / 14;
    int j = (int)(r % Ff); int b = (int)(r / Ff);
    gadT[e] = (j < Nn) ? gad[((long)(b * 14 + c)) * Nn + j]
                       : gaddb[((long)(b * 14 + c)) * Nn + (j - Nn)];
  }
}

// ---------------------------------------------------------------------------
// Kernel 2: brute-force 3D KNN (top-16 smallest sq dist over 8192 candidates).
// One thread = one query point; candidates tiled through LDS.
// ---------------------------------------------------------------------------
__global__ void __launch_bounds__(128)
k_knn(const float* __restrict__ gadT, int* __restrict__ idxOut) {
  int n = blockIdx.x * blockDim.x + threadIdx.x;
  int b = blockIdx.y;
  const float* basev = gadT + (size_t)b * Ff * 14;
  float sx = basev[(size_t)n * 14 + 0];
  float sy = basev[(size_t)n * 14 + 1];
  float sz = basev[(size_t)n * 14 + 2];
  __shared__ float cx[128], cy[128], cz[128];
  float bd[16]; int bi[16];
#pragma unroll
  for (int t = 0; t < 16; ++t) { bd[t] = 3.4e38f; bi[t] = 0; }
  for (int tile = 0; tile < Ff / 128; ++tile) {
    const float* cp = basev + (size_t)(tile * 128 + threadIdx.x) * 14;
    cx[threadIdx.x] = cp[0]; cy[threadIdx.x] = cp[1]; cz[threadIdx.x] = cp[2];
    if (tile + 1 < Ff / 128)
      __builtin_prefetch(basev + (size_t)((tile + 1) * 128 + threadIdx.x) * 14, 0, 0);
    __syncthreads();
#pragma unroll 4
    for (int jj = 0; jj < 128; ++jj) {
      float dx = sx - cx[jj], dy = sy - cy[jj], dz = sz - cz[jj];
      float d = dx * dx + dy * dy + dz * dz;
      if (d < bd[15]) {
        bd[15] = d; bi[15] = tile * 128 + jj;
#pragma unroll
        for (int t = 15; t > 0; --t) {
          if (bd[t] < bd[t - 1]) {
            float td = bd[t]; bd[t] = bd[t - 1]; bd[t - 1] = td;
            int   ti = bi[t]; bi[t] = bi[t - 1]; bi[t - 1] = ti;
          }
        }
      }
    }
    __syncthreads();
  }
  int* op = idxOut + ((size_t)(b * Nn + n)) * Kn;
#pragma unroll
  for (int t = 0; t < 16; ++t) op[t] = bi[t];
}

// ---------------------------------------------------------------------------
// Kernel 3: pos conv1 (64x14 @ 14x16 per point, tiny -> VALU) + global stats.
// ---------------------------------------------------------------------------
__global__ void __launch_bounds__(256)
k_pos1(const float* __restrict__ gadT, const int* __restrict__ idxIn,
       const float* __restrict__ pw1, const float* __restrict__ pb1,
       _Float16* __restrict__ pospre, float* __restrict__ psum, float* __restrict__ psq) {
  int pt = blockIdx.x; int b = pt >> 12; int n = pt & (Nn - 1);
  int t = threadIdx.x;
  __shared__ int   sIdx[16];
  __shared__ float gg[14][16];
  __shared__ float pgv[14];
  __shared__ float sSum[64], sSq[64];
  if (t < 16) sIdx[t] = idxIn[(size_t)pt * Kn + t];
  if (t < 64) { sSum[t] = 0.f; sSq[t] = 0.f; }
  __syncthreads();
  if (t < 224) { int c = t >> 4, k = t & 15;
    gg[c][k] = gadT[((size_t)b * Ff + sIdx[k]) * 14 + c]; }
  if (t >= 224 && t < 238) { int c = t - 224;
    pgv[c] = gadT[((size_t)b * Ff + n) * 14 + c]; }
  __syncthreads();
  int c = t & 63, kb = (t >> 6) * 4;
  float ls = 0.f, lq = 0.f;
#pragma unroll
  for (int kk = 0; kk < 4; ++kk) {
    int k = kb + kk;
    float acc = pb1[c];
#pragma unroll
    for (int j = 0; j < 14; ++j) acc += pw1[c * 14 + j] * (pgv[j] - gg[j][k]);
    pospre[((size_t)pt * PH + c) * Kn + k] = (_Float16)acc;
    ls += acc; lq += acc * acc;
  }
  atomicAdd(&sSum[c], ls); atomicAdd(&sSq[c], lq);
  __syncthreads();
  if (t < 64) { atomicAdd(&psum[t], sSum[t]); atomicAdd(&psq[t], sSq[t]); }
}

// ---------------------------------------------------------------------------
// Kernel 4: finalize batchnorm stats -> mean, invstd.
// ---------------------------------------------------------------------------
__global__ void k_finalize(const float* __restrict__ sum, const float* __restrict__ sq,
                           float* __restrict__ mean, float* __restrict__ istd,
                           int Cn, float invCnt) {
  int t = blockIdx.x * blockDim.x + threadIdx.x;
  if (t < Cn) {
    float m = sum[t] * invCnt;
    float v = sq[t] * invCnt - m * m;
    mean[t] = m;
    istd[t] = rsqrtf(v + EPSf);
  }
}

// ---------------------------------------------------------------------------
// Kernel 5 (WMMA): per point -- bn/relu(pos_pre), pos_emb = w_pos2 @ h (WMMA),
// x = qk_rel + pos_emb, a_pre = w_attn1 @ x (WMMA) used only for stats.
// Block = 128 threads (4 waves), PPB points/block.
// w_attn1 (128 KB fp16) is staged in dynamic LDS once per block; w_pos2
// fragments are register-cached across the whole point loop.
// ---------------------------------------------------------------------------
__global__ void __launch_bounds__(128)
k_pos2_attn1(const float* __restrict__ featT, const float* __restrict__ feat,
             const int* __restrict__ idxIn, const _Float16* __restrict__ pospre,
             const _Float16* __restrict__ wpos2h, const float* __restrict__ pb2,
             const _Float16* __restrict__ wattn1h, const float* __restrict__ ab1,
             const float* __restrict__ pmean, const float* __restrict__ pistd,
             const float* __restrict__ pg1, const float* __restrict__ pbe1,
             _Float16* __restrict__ posemb, float* __restrict__ asum, float* __restrict__ asq) {
  extern __shared__ _Float16 dynw[];                  // [AH*Cc] staged w_attn1
  __shared__ int   sIdx[16];
  __shared__ float featc[128];
  __shared__ float gf[16][128];                       // gathered neighbor features (f32)
  __shared__ __align__(32) _Float16 hT[16][64];       // B-panel for WMMA1
  __shared__ __align__(32) _Float16 xT[16][128];      // B-panel for WMMA2
  __shared__ float aSum[512], aSq[512];
  int t = threadIdx.x;
  int wave = t >> 5, lane = t & 31, hi = (lane >= 16), ln = lane & 15;

  // Stage w_attn1 into LDS (8192 x 16B) and zero block stats.
  {
    v8h* dst = (v8h*)dynw;
    const v8h* src = (const v8h*)wattn1h;
    for (int i = t; i < (AH * Cc) / 8; i += 128) dst[i] = src[i];
  }
#pragma unroll
  for (int i = 0; i < 4; ++i) { aSum[t + 128 * i] = 0.f; aSq[t + 128 * i] = 0.f; }

  // Register-cache the 4 w_pos2 A-fragments this wave needs (invariant).
  v16h aPos[2][2];
#pragma unroll
  for (int rt2 = 0; rt2 < 2; ++rt2)
#pragma unroll
    for (int ks = 0; ks < 2; ++ks) {
      int rt = wave * 2 + rt2;
      int kb = ks * 32 + (hi ? 16 : 0);
      aPos[rt2][ks] = load_a_frag(wpos2h + (size_t)(rt * 16 + ln) * PH, kb);
    }
  __syncthreads();

  for (int pp = 0; pp < PPB; ++pp) {
    int pt = blockIdx.x * PPB + pp;
    int b = pt >> 12, n = pt & (Nn - 1);
    if (t < 16) sIdx[t] = idxIn[(size_t)pt * Kn + t];
    featc[t] = feat[((size_t)(b * Cc + t)) * Nn + n];
    __syncthreads();
    for (int k = 0; k < 16; ++k)
      gf[k][t] = featT[((size_t)b * Ff + sIdx[k]) * Cc + t];   // contiguous per k
    // pos_pre -> bn -> relu -> hT (transposed B-panel layout)
#pragma unroll
    for (int j = 0; j < 8; ++j) {
      int e = t * 8 + j; int c = e >> 4, k = e & 15;
      float v = (float)pospre[((size_t)pt * PH + c) * Kn + k];
      v = (v - pmean[c]) * pistd[c] * pg1[c] + pbe1[c];
      hT[k][c] = (_Float16)fmaxf(v, 0.f);
    }
    __syncthreads();
    // WMMA1: pos_emb(128x16) = w_pos2(128x64) @ h(64x16); 2 row tiles / wave.
#pragma unroll
    for (int rt2 = 0; rt2 < 2; ++rt2) {
      int rt = wave * 2 + rt2;
      v8f acc = {};
#pragma unroll
      for (int ks = 0; ks < 2; ++ks) {
        v16h Bf = *(const v16h*)(&hT[ln][ks * 32 + (hi ? 16 : 0)]);
        acc = wmma_f16(aPos[rt2][ks], Bf, acc);
      }
#pragma unroll
      for (int r = 0; r < 8; ++r) {
        int row = rt * 16 + r + (hi ? 8 : 0);
        float pe = acc[r] + pb2[row];
        posemb[((size_t)pt * Cc + row) * Kn + ln] = (_Float16)pe;
        xT[ln][row] = (_Float16)(featc[row] - gf[ln][row] + pe);
      }
    }
    __syncthreads();
    // WMMA2: a_pre(512x16) = w_attn1(512x128) @ x(128x16); stats only.
#pragma unroll
    for (int i8 = 0; i8 < 8; ++i8) {
      int rt = wave + i8 * 4;
      v8f acc = {};
#pragma unroll
      for (int ks = 0; ks < 4; ++ks) {
        int kb = ks * 32 + (hi ? 16 : 0);
        v16h A  = load_a_frag(dynw + (size_t)(rt * 16 + ln) * Cc, kb);   // from LDS
        v16h Bf = *(const v16h*)(&xT[ln][ks * 32 + (hi ? 16 : 0)]);
        acc = wmma_f16(A, Bf, acc);
      }
#pragma unroll
      for (int r = 0; r < 8; ++r) {
        int row = rt * 16 + r + (hi ? 8 : 0);
        float v = acc[r] + ab1[row];
        float s = v, q = v * v;
        s += __shfl_xor(s, 1); s += __shfl_xor(s, 2);
        s += __shfl_xor(s, 4); s += __shfl_xor(s, 8);
        q += __shfl_xor(q, 1); q += __shfl_xor(q, 2);
        q += __shfl_xor(q, 4); q += __shfl_xor(q, 8);
        if (ln == 0) { atomicAdd(&aSum[row], s); atomicAdd(&aSq[row], q); }
      }
    }
    __syncthreads();
  }
#pragma unroll
  for (int i = 0; i < 4; ++i) {
    int c = t + 128 * i;
    atomicAdd(&asum[c], aSum[c]); atomicAdd(&asq[c], aSq[c]);
  }
}

// ---------------------------------------------------------------------------
// Kernel 6 (WMMA): recompute a_pre (cheaper than storing 1 GB), bn/relu,
// logits = w_attn2 @ a (WMMA), softmax over K=16 (cross-lane), weighted sum.
// Both w_attn1 and w_attn2 (256 KB fp16) staged in dynamic LDS per block
// (fits the 320 KB/workgroup CDNA5 LDS budget together with ~25 KB static).
// ---------------------------------------------------------------------------
__global__ void __launch_bounds__(128)
k_attn2_final(const float* __restrict__ featT, const float* __restrict__ feat,
              const int* __restrict__ idxIn, const _Float16* __restrict__ posemb,
              const _Float16* __restrict__ wattn1h, const float* __restrict__ ab1,
              const _Float16* __restrict__ wattn2h, const float* __restrict__ ab2,
              const float* __restrict__ amean, const float* __restrict__ aistd,
              const float* __restrict__ ag1, const float* __restrict__ abe1,
              float* __restrict__ outp) {
  extern __shared__ _Float16 dynw[];               // [2*AH*Cc]: w_attn1 | w_attn2
  __shared__ int   sIdx[16];
  __shared__ float featc[128];
  __shared__ __align__(32) _Float16 xT[16][128];   // qk_rel + pos_emb
  __shared__ __align__(32) _Float16 sT[16][128];   // group_feat + pos_emb
  __shared__ __align__(32) _Float16 aT[16][512];   // relu(bn(a_pre)) B-panel
  int t = threadIdx.x;
  int wave = t >> 5, lane = t & 31, hi = (lane >= 16), ln = lane & 15;

  {
    v8h* dst = (v8h*)dynw;
    const v8h* s1 = (const v8h*)wattn1h;
    const v8h* s2 = (const v8h*)wattn2h;
    for (int i = t; i < (AH * Cc) / 8; i += 128) dst[i] = s1[i];
    for (int i = t; i < (AH * Cc) / 8; i += 128) dst[(AH * Cc) / 8 + i] = s2[i];
  }
  __syncthreads();
  const _Float16* wa1 = dynw;
  const _Float16* wa2 = dynw + AH * Cc;

  for (int pp = 0; pp < PPB; ++pp) {
    int pt = blockIdx.x * PPB + pp;
    int b = pt >> 12, n = pt & (Nn - 1);
    if (t < 16) sIdx[t] = idxIn[(size_t)pt * Kn + t];
    featc[t] = feat[((size_t)(b * Cc + t)) * Nn + n];
    __syncthreads();
#pragma unroll
    for (int k = 0; k < 16; ++k) {
      float g  = featT[((size_t)b * Ff + sIdx[k]) * Cc + t];
      float pe = (float)posemb[((size_t)pt * Cc + t) * Kn + k];
      xT[k][t] = (_Float16)(featc[t] - g + pe);
      sT[k][t] = (_Float16)(g + pe);
    }
    __syncthreads();
    // a(512x16) = relu(bn(w_attn1 @ x))
#pragma unroll
    for (int i8 = 0; i8 < 8; ++i8) {
      int rt = wave + i8 * 4;
      v8f acc = {};
#pragma unroll
      for (int ks = 0; ks < 4; ++ks) {
        int kb = ks * 32 + (hi ? 16 : 0);
        v16h A  = load_a_frag(wa1 + (size_t)(rt * 16 + ln) * Cc, kb);
        v16h Bf = *(const v16h*)(&xT[ln][ks * 32 + (hi ? 16 : 0)]);
        acc = wmma_f16(A, Bf, acc);
      }
#pragma unroll
      for (int r = 0; r < 8; ++r) {
        int row = rt * 16 + r + (hi ? 8 : 0);
        float v = acc[r] + ab1[row];
        v = (v - amean[row]) * aistd[row] * ag1[row] + abe1[row];
        aT[ln][row] = (_Float16)fmaxf(v, 0.f);
      }
    }
    __syncthreads();
    // logits(128x16) = w_attn2(128x512) @ a(512x16); softmax over cols; reduce.
#pragma unroll
    for (int rtw = 0; rtw < 2; ++rtw) {
      int rt = wave + rtw * 4;
      v8f acc = {};
#pragma unroll
      for (int ks = 0; ks < 16; ++ks) {
        int kb = ks * 32 + (hi ? 16 : 0);
        v16h A  = load_a_frag(wa2 + (size_t)(rt * 16 + ln) * AH, kb);
        v16h Bf = *(const v16h*)(&aT[ln][ks * 32 + (hi ? 16 : 0)]);
        acc = wmma_f16(A, Bf, acc);
      }
#pragma unroll
      for (int r = 0; r < 8; ++r) {
        int row = rt * 16 + r + (hi ? 8 : 0);
        float v = acc[r] + ab2[row];
        float mx = v;
        mx = fmaxf(mx, __shfl_xor(mx, 1)); mx = fmaxf(mx, __shfl_xor(mx, 2));
        mx = fmaxf(mx, __shfl_xor(mx, 4)); mx = fmaxf(mx, __shfl_xor(mx, 8));
        float e = __expf(v - mx);
        float ssum = e;
        ssum += __shfl_xor(ssum, 1); ssum += __shfl_xor(ssum, 2);
        ssum += __shfl_xor(ssum, 4); ssum += __shfl_xor(ssum, 8);
        float contrib = (e / ssum) * (float)sT[ln][row];
        contrib += __shfl_xor(contrib, 1); contrib += __shfl_xor(contrib, 2);
        contrib += __shfl_xor(contrib, 4); contrib += __shfl_xor(contrib, 8);
        if (ln == 0) outp[((size_t)(b * Cc + row)) * Nn + n] = contrib;
      }
    }
    __syncthreads();
  }
}

// ---------------------------------------------------------------------------
extern "C" void kernel_launch(void* const* d_in, const int* in_sizes, int n_in,
                              void* d_out, int out_size, void* d_ws, size_t ws_size,
                              hipStream_t stream) {
  const float* pcd_gad    = (const float*)d_in[0];
  const float* feat       = (const float*)d_in[1];
  const float* pcd_fb_gad = (const float*)d_in[2];
  const float* feat_fb    = (const float*)d_in[3];
  const float* pos_w1     = (const float*)d_in[4];
  const float* pos_b1     = (const float*)d_in[5];
  const float* pos_g1     = (const float*)d_in[6];
  const float* pos_be1    = (const float*)d_in[7];
  const float* pos_w2     = (const float*)d_in[8];
  const float* pos_b2     = (const float*)d_in[9];
  const float* attn_w1    = (const float*)d_in[10];
  const float* attn_b1    = (const float*)d_in[11];
  const float* attn_g1    = (const float*)d_in[12];
  const float* attn_be1   = (const float*)d_in[13];
  const float* attn_w2    = (const float*)d_in[14];
  const float* attn_b2    = (const float*)d_in[15];

  char* base = (char*)d_ws;
  size_t off = 0;
  auto take = [&](size_t bytes) -> void* {
    void* p = base + off;
    off = (off + bytes + 255) & ~(size_t)255;
    return p;
  };
  int*      idx     = (int*)      take((size_t)NPTS * Kn * sizeof(int));
  float*    featT   = (float*)    take((size_t)Bb * Ff * Cc * sizeof(float));
  float*    gadT    = (float*)    take((size_t)Bb * Ff * 14 * sizeof(float));
  _Float16* wpos2h  = (_Float16*) take((size_t)Cc * PH * 2);
  _Float16* wattn1h = (_Float16*) take((size_t)AH * Cc * 2);
  _Float16* wattn2h = (_Float16*) take((size_t)Cc * AH * 2);
  _Float16* pospre  = (_Float16*) take((size_t)NPTS * PH * Kn * 2);
  _Float16* posemb  = (_Float16*) take((size_t)NPTS * Cc * Kn * 2);
  float*    stats   = (float*)    take(2304 * sizeof(float));
  float *psum = stats,       *psq = stats + 64,  *pmean = stats + 128,  *pistd = stats + 192;
  float *asum = stats + 256, *asq = stats + 768, *amean = stats + 1280, *aistd = stats + 1792;

  const float invCnt = 1.f / ((float)NPTS * (float)Kn);

  k_prep<<<(AH * Cc + 255) / 256, 256, 0, stream>>>(
      pos_w2, attn_w1, attn_w2, wpos2h, wattn1h, wattn2h, stats);
  {
    long tot = (long)Bb * Ff * Cc;
    k_transpose<<<(unsigned)((tot + 255) / 256), 256, 0, stream>>>(
        feat, feat_fb, pcd_gad, pcd_fb_gad, featT, gadT);
  }
  k_knn<<<dim3(Nn / 128, Bb), 128, 0, stream>>>(gadT, idx);
  k_pos1<<<NPTS, 256, 0, stream>>>(gadT, idx, pos_w1, pos_b1, pospre, psum, psq);
  k_finalize<<<1, 64, 0, stream>>>(psum, psq, pmean, pistd, 64, invCnt);
  k_pos2_attn1<<<NPTS / PPB, 128, (size_t)AH * Cc * 2, stream>>>(
      featT, feat, idx, pospre, wpos2h, pos_b2, wattn1h, attn_b1,
      pmean, pistd, pos_g1, pos_be1, posemb, asum, asq);
  k_finalize<<<1, 512, 0, stream>>>(asum, asq, amean, aistd, 512, invCnt);
  k_attn2_final<<<NPTS / PPB, 128, (size_t)2 * AH * Cc * 2, stream>>>(
      featT, feat, idx, posemb, wattn1h, attn_b1, wattn2h, attn_b2,
      amean, aistd, attn_g1, attn_be1, (float*)d_out);

  (void)in_sizes; (void)n_in; (void)out_size; (void)ws_size;
}
